// SemiMarkovModule_87548613361998
// MI455X (gfx1250) — matrix operand order repaired and checked by reading, not compiled, not run on hardware
//
#include <hip/hip_runtime.h>
#include <hip/hip_bf16.h>
#include <math.h>

// ---------------------------------------------------------------------------
// Semi-Markov HMM logZ for MI455X (gfx1250, wave32).
// Phase 1 (emission): bandwidth-bound GEMM -> v_wmma_f32_16x16x32_f16 with
//   hi/lo f16 split of the features operand (near-fp32 cross term). NT is a
//   template parameter so accumulators live in fixed registers (no copies).
// Phase 2 (scan): latency-bound recurrence -> 1 WG per batch chain, LDS ring
//   buffers (column-private, barrier-free), padded transition matrix to kill
//   64-way bank conflicts, exp2/log2 LSE, global prefetch of emission stream.
// ---------------------------------------------------------------------------

typedef __attribute__((ext_vector_type(16))) _Float16 v16h;
typedef __attribute__((ext_vector_type(8)))  float    v8f;

#define BIG_NEG (-1000000000.0f)
#define KCAP 32
#define CCAP 64
#define LOG2E 1.4426950408889634f
#define LN2   0.6931471805599453f

// ------------------------------ prep ---------------------------------------
__global__ void smm_prep(const float* __restrict__ means, const float* __restrict__ cov,
                         const float* __restrict__ tl,    const float* __restrict__ il,
                         const float* __restrict__ plr,   const int* __restrict__ maxk_p,
                         int C, int D,
                         _Float16* __restrict__ mw, float* __restrict__ m2,
                         float* __restrict__ trans, float* __restrict__ initsm,
                         float* __restrict__ lenlp, float* __restrict__ consts)
{
    const int tid = threadIdx.x;
    int K = *maxk_p; if (K > KCAP) K = KCAP; if (K < 1) K = 1;

    // B operand of the emission GEMM: (means / var) in f16, row-major (C x D)
    for (int idx = tid; idx < C * D; idx += blockDim.x) {
        int d = idx % D;
        float var = cov[(size_t)d * D + d];
        mw[idx] = (_Float16)(means[idx] / var);
    }
    // m2[c] = sum_d means^2 / var  (exact fp32)
    for (int c = tid; c < C; c += blockDim.x) {
        float s = 0.f;
        for (int d = 0; d < D; ++d) {
            float var = cov[(size_t)d * D + d];
            float m   = means[(size_t)c * D + d];
            s += m * m / var;
        }
        m2[c] = s;
    }
    // transition: mask diagonal, log_softmax over axis 0 (per column j)
    for (int j = tid; j < C; j += blockDim.x) {
        float mx = BIG_NEG;
        for (int i = 0; i < C; ++i) {
            float v = (i == j) ? BIG_NEG : tl[(size_t)i * C + j];
            mx = fmaxf(mx, v);
        }
        float s = 0.f;
        for (int i = 0; i < C; ++i) {
            float v = (i == j) ? BIG_NEG : tl[(size_t)i * C + j];
            s += exp2f((v - mx) * LOG2E);
        }
        float lz = mx + log2f(s) * LN2;
        for (int i = 0; i < C; ++i) {
            float v = (i == j) ? BIG_NEG : tl[(size_t)i * C + j];
            trans[(size_t)i * C + j] = v - lz;
        }
    }
    // init log-softmax + constant term (logdet + D*log(2pi)) : tiny, thread 0
    if (tid == 0) {
        float mx = BIG_NEG;
        for (int c = 0; c < C; ++c) mx = fmaxf(mx, il[c]);
        float s = 0.f;
        for (int c = 0; c < C; ++c) s += exp2f((il[c] - mx) * LOG2E);
        float lz = mx + log2f(s) * LN2;
        for (int c = 0; c < C; ++c) initsm[c] = il[c] - lz;
        float ld = 0.f;
        for (int d = 0; d < D; ++d) ld += logf(cov[(size_t)d * D + d]);
        consts[0] = ld + (float)D * logf(2.0f * (float)M_PI);
    }
    // length log-probs: len_lp[k][c] = (k+1)*lr - exp(lr) - lgamma(k+2)
    for (int idx = tid; idx < K * C; idx += blockDim.x) {
        int k = idx / C, c = idx % C;
        float lr = plr[c];
        lenlp[idx] = (float)(k + 1) * lr - exp2f(lr * LOG2E) - lgammaf((float)(k + 2));
    }
}

// ---------------------------- emission (WMMA) ------------------------------
// Grid: (BN+127)/128 blocks x 256 threads. Each wave owns a 16-row stripe and
// NT 16-wide column tiles (NT = C/16, compile-time). A is split hi/lo f16 so
// the cross term is near-fp32 while x^2 stays exact fp32 VALU.
template<int NT>
__global__ __launch_bounds__(256) void smm_emission_t(
    const float* __restrict__ feat, const _Float16* __restrict__ mw,
    const float* __restrict__ m2,   const float* __restrict__ consts,
    float* __restrict__ emis, int BN, int D, int C)
{
    const int lane    = threadIdx.x & 31;
    const int wave    = threadIdx.x >> 5;
    const int half_id = lane >> 4;        // wave32 half
    const int mrow    = lane & 15;
    const int row0    = blockIdx.x * 128 + wave * 16;
    const float cterm = consts[0];

    v8f acc[NT];
    #pragma unroll
    for (int nt = 0; nt < NT; ++nt) acc[nt] = (v8f){};

    int row = row0 + mrow; if (row >= BN) row = BN - 1;
    const float* arow = feat + (size_t)row * D;
    const _Float16* brow = mw + (size_t)mrow * D + (half_id ? 16 : 0);

    float x2p = 0.f;
    for (int k0 = 0; k0 < D; k0 += 32) {
        // --- issue all B loads first (one clause) so the f32->f16 conversion
        //     VALU work below hides their latency.
        // B fragment, contiguous K order (matches A): lanes 0-15 hold K=k0+[0..15]
        // of column n, lanes 16-31 hold K=k0+[16..31].
        v16h bf[NT];
        #pragma unroll
        for (int nt = 0; nt < NT; ++nt)
            bf[nt] = *(const v16h*)(brow + (size_t)(nt * 16) * D + k0);

        // --- A fragment, contiguous K order (matches B):
        // lanes 0-15 : K = k0+[0..15]
        // lanes 16-31: K = k0+[16..31]
        const int ka = k0 + (half_id ? 16 : 0);
        const int kb = ka + 8;
        float av[16];
        {
            const float4 a0 = *(const float4*)(arow + ka);
            const float4 a1 = *(const float4*)(arow + ka + 4);
            const float4 b0 = *(const float4*)(arow + kb);
            const float4 b1 = *(const float4*)(arow + kb + 4);
            av[0]=a0.x;  av[1]=a0.y;  av[2]=a0.z;  av[3]=a0.w;
            av[4]=a1.x;  av[5]=a1.y;  av[6]=a1.z;  av[7]=a1.w;
            av[8]=b0.x;  av[9]=b0.y;  av[10]=b0.z; av[11]=b0.w;
            av[12]=b1.x; av[13]=b1.y; av[14]=b1.z; av[15]=b1.w;
        }
        v16h ahi, alo;
        #pragma unroll
        for (int i = 0; i < 16; ++i) {
            float x = av[i];
            _Float16 h = (_Float16)x;
            ahi[i] = h;
            alo[i] = (_Float16)(x - (float)h);
            x2p += x * x;
        }
        #pragma unroll
        for (int nt = 0; nt < NT; ++nt) {
            acc[nt] = __builtin_amdgcn_wmma_f32_16x16x32_f16(
                false, ahi, false, bf[nt], (short)0, acc[nt], false, false);
            acc[nt] = __builtin_amdgcn_wmma_f32_16x16x32_f16(
                false, alo, false, bf[nt], (short)0, acc[nt], false, false);
        }
    }

    // reduce per-row x^2 partials (row m lives in lanes m and m+16)
    __shared__ float x2s[8][32];
    x2s[wave][lane] = x2p;
    __syncthreads();

    // C/D layout: lanes 0-15 -> N=lane, M=r ; lanes 16-31 -> N=lane-16, M=8+r
    const int mbase = half_id * 8;
    #pragma unroll
    for (int nt = 0; nt < NT; ++nt) {
        const int col = nt * 16 + mrow;
        const float mm2 = m2[col];
        #pragma unroll
        for (int r = 0; r < 8; ++r) {
            const int mr = mbase + r;
            const int orow = row0 + mr;
            if (orow < BN) {
                float x2v = x2s[wave][mr] + x2s[wave][16 + mr];
                float val = -0.5f * (x2v - 2.0f * acc[nt][r] + mm2 + cterm);
                emis[(size_t)orow * C + col] = val;
            }
        }
    }
}

// ------------------------------- scan --------------------------------------
// One workgroup per batch chain (C threads, thread c owns state column c).
// beta/CE ring buffers are column-private in LDS -> no barrier needed for
// them; only alpha_sh needs the 2 barriers per step. trans_l is padded to
// stride C+1 so the per-j row reads (stride 64 dwords otherwise) do not all
// land in one LDS bank.
__global__ __launch_bounds__(CCAP) void smm_scan(
    const float* __restrict__ emis, const int* __restrict__ lengths,
    const float* __restrict__ trans, const float* __restrict__ initsm,
    const float* __restrict__ lenlp, const int* __restrict__ maxk_p,
    float* __restrict__ out, int B, int N, int C)
{
    __shared__ float beta_ring[KCAP][CCAP];
    __shared__ float ce_ring[KCAP + 1][CCAP];
    __shared__ float alpha_sh[CCAP];
    __shared__ float trans_l[CCAP][CCAP + 1];   // +1 pad: conflict-free rows
    __shared__ float len_l[KCAP][CCAP];
    __shared__ float sv[CCAP];
    __shared__ float red[CCAP];

    const int b = blockIdx.x;
    const int c = threadIdx.x;
    int K = *maxk_p; if (K > KCAP) K = KCAP; if (K < 1) K = 1;
    const int Lb = lengths[b];

    for (int j = 0; j < C; ++j) trans_l[c][j] = trans[(size_t)c * C + j];
    for (int k = 0; k < K; ++k) {
        len_l[k][c]     = lenlp[(size_t)k * C + c];   // len_lp[duration k+1]
        beta_ring[k][c] = BIG_NEG;
    }
    for (int k = 0; k <= K; ++k) ce_ring[k][c] = 0.f;
    beta_ring[0][c] = initsm[c];                       // beta_0 = init
    sv[c] = BIG_NEG;
    float ce_reg = 0.f;
    __syncthreads();

    const float* erow = emis + (size_t)b * N * C + c;
    // ring slot of time (t-K) rotates by +1 each step; precompute bases for
    // t=1: tau0 = 1-K  ->  beta slot (tau0 mod K), ce slot (tau0 mod (K+1)).
    int sb_base = (1 - K) % K;       if (sb_base < 0) sb_base += K;
    int sc_base = (1 - K) % (K + 1); if (sc_base < 0) sc_base += (K + 1);
    int wslot_b = 1 % K;             // beta write slot at t=1
    int wslot_c = 1 % (K + 1);       // ce write slot at t=1

    for (int t = 1; t <= N; ++t) {
        float e = erow[(size_t)(t - 1) * C];
        if (t + 7 < N) __builtin_prefetch(erow + (size_t)(t + 7) * C, 0, 1);
        ce_reg += e;
        ce_ring[wslot_c][c] = ce_reg;   // slot of t-(K+1): not read this step

        // alpha_t[c] = LSE_j( beta_{t-K+j} + (CE_t - CE_{t-K+j}) + len[K-1-j] )
        float mx = BIG_NEG;
        {
            int sb = sb_base, sc = sc_base;
            for (int j = 0; j < K; ++j) {
                float v = beta_ring[sb][c] + (ce_reg - ce_ring[sc][c])
                        + len_l[K - 1 - j][c];
                mx = fmaxf(mx, v);
                if (++sb == K) sb = 0;
                if (++sc == K + 1) sc = 0;
            }
        }
        float s = 0.f;
        {
            int sb = sb_base, sc = sc_base;
            for (int j = 0; j < K; ++j) {
                float v = beta_ring[sb][c] + (ce_reg - ce_ring[sc][c])
                        + len_l[K - 1 - j][c];
                s += exp2f((v - mx) * LOG2E);
                if (++sb == K) sb = 0;
                if (++sc == K + 1) sc = 0;
            }
        }
        float alpha = mx + log2f(s) * LN2;
        alpha_sh[c] = alpha;
        if (t == Lb) sv[c] = alpha;
        __syncthreads();

        // beta_t[c] = LSE_j( alpha[j] + trans[c][j] )
        float mx2 = BIG_NEG;
        for (int j = 0; j < C; ++j) mx2 = fmaxf(mx2, alpha_sh[j] + trans_l[c][j]);
        float s2 = 0.f;
        for (int j = 0; j < C; ++j)
            s2 += exp2f((alpha_sh[j] + trans_l[c][j] - mx2) * LOG2E);
        beta_ring[wslot_b][c] = mx2 + log2f(s2) * LN2;  // column-private write
        __syncthreads();                                 // protects alpha_sh

        if (++sb_base == K) sb_base = 0;
        if (++sc_base == K + 1) sc_base = 0;
        if (++wslot_b == K) wslot_b = 0;
        if (++wslot_c == K + 1) wslot_c = 0;
    }

    // out[b] = LSE_c sv[c]
    red[c] = sv[c];
    __syncthreads();
    for (int off = C >> 1; off > 0; off >>= 1) {
        if (c < off) red[c] = fmaxf(red[c], red[c + off]);
        __syncthreads();
    }
    float m = red[0];
    __syncthreads();
    red[c] = exp2f((sv[c] - m) * LOG2E);
    __syncthreads();
    for (int off = C >> 1; off > 0; off >>= 1) {
        if (c < off) red[c] += red[c + off];
        __syncthreads();
    }
    if (c == 0) out[b] = m + log2f(red[0]) * LN2;
}

// ------------------------------ launcher -----------------------------------
extern "C" void kernel_launch(void* const* d_in, const int* in_sizes, int n_in,
                              void* d_out, int out_size, void* d_ws, size_t ws_size,
                              hipStream_t stream)
{
    const float* feat    = (const float*)d_in[0];
    const int*   lengths = (const int*)  d_in[1];
    const float* means   = (const float*)d_in[2];
    const float* cov     = (const float*)d_in[3];
    const float* tl      = (const float*)d_in[4];
    const float* il      = (const float*)d_in[5];
    const float* plr     = (const float*)d_in[6];
    const int*   maxk    = (const int*)  d_in[7];

    const int B  = in_sizes[1];
    const int C  = in_sizes[5];
    const int D  = in_sizes[2] / C;
    const int N  = in_sizes[0] / (B * D);
    const int BN = B * N;

    char* ws = (char*)d_ws;
    size_t off = 0;
    auto take = [&](size_t bytes) -> void* {
        void* p = (void*)(ws + off);
        off += (bytes + 255) & ~(size_t)255;
        return p;
    };
    float*    emis   = (float*)   take((size_t)BN * C * sizeof(float));
    _Float16* mw     = (_Float16*)take((size_t)C * D * sizeof(_Float16));
    float*    m2     = (float*)   take((size_t)C * sizeof(float));
    float*    trans  = (float*)   take((size_t)C * C * sizeof(float));
    float*    initsm = (float*)   take((size_t)C * sizeof(float));
    float*    lenlp  = (float*)   take((size_t)KCAP * C * sizeof(float));
    float*    consts = (float*)   take(4 * sizeof(float));
    (void)n_in; (void)out_size; (void)ws_size;

    smm_prep<<<1, 256, 0, stream>>>(means, cov, tl, il, plr, maxk, C, D,
                                    mw, m2, trans, initsm, lenlp, consts);

    const int nblk = (BN + 127) / 128;
    switch (C >> 4) {                 // NT = C/16 (C=64 -> 4)
    case 1:  smm_emission_t<1><<<nblk, 256, 0, stream>>>(feat, mw, m2, consts, emis, BN, D, C); break;
    case 2:  smm_emission_t<2><<<nblk, 256, 0, stream>>>(feat, mw, m2, consts, emis, BN, D, C); break;
    case 3:  smm_emission_t<3><<<nblk, 256, 0, stream>>>(feat, mw, m2, consts, emis, BN, D, C); break;
    default: smm_emission_t<4><<<nblk, 256, 0, stream>>>(feat, mw, m2, consts, emis, BN, D, C); break;
    }

    smm_scan<<<B, C, 0, stream>>>(emis, lengths, trans, initsm, lenlp, maxk,
                                  (float*)d_out, B, N, C);
}
